// Transformer_13898514170127
// MI455X (gfx1250) — matrix operand order repaired and checked by reading, not compiled, not run on hardware
//
#include <hip/hip_runtime.h>
#include <hip/hip_bf16.h>

typedef __attribute__((ext_vector_type(16))) _Float16 v16h;
typedef __attribute__((ext_vector_type(8)))  float    v8f;
typedef __attribute__((ext_vector_type(4)))  int      v4i;

#define Dm   768
#define Tm   1024
#define Bm   4
#define Hm   4
#define DHm  192
#define Lm   12
#define Vm   50257
#define ROWS (Bm * Tm)           // 4096

#define AS1 __attribute__((address_space(1)))
#define AS3 __attribute__((address_space(3)))

#if defined(__AMDGCN__) && __has_builtin(__builtin_amdgcn_global_load_async_to_lds_b128)
#define HAVE_ASYNC 1
#else
#define HAVE_ASYNC 0
#endif

#if HAVE_ASYNC
__device__ __forceinline__ void async_copy16B(const _Float16* g, _Float16* l)
{
    __builtin_amdgcn_global_load_async_to_lds_b128(
        (AS1 v4i*)g, (AS3 v4i*)l, /*offset=*/0, /*cpol=*/0);
}
template<int N>
__device__ __forceinline__ void wait_async()
{
#if __has_builtin(__builtin_amdgcn_s_wait_asynccnt)
    __builtin_amdgcn_s_wait_asynccnt(N);
#else
    asm volatile("s_wait_asynccnt %0" :: "n"(N) : "memory");
#endif
}
#endif

// ---------------------------------------------------------------------------
// Embedding: h[b,t,:] = tok_emb[x[b,t],:] + pos_emb[t,:]
// ---------------------------------------------------------------------------
__global__ void embed_kernel(const int* __restrict__ x,
                             const float* __restrict__ tok,
                             const float* __restrict__ pos,
                             float* __restrict__ h)
{
    const int bt = blockIdx.x;              // 0..4095
    const int t  = bt & (Tm - 1);
    const int id = x[bt];
    const float* tr = tok + (size_t)id * Dm;
    const float* pr = pos + (size_t)t  * Dm;
    float* hr = h + (size_t)bt * Dm;
    for (int i = threadIdx.x; i < Dm; i += blockDim.x)
        hr[i] = tr[i] + pr[i];
}

// ---------------------------------------------------------------------------
// LayerNorm over D=768, writes f16 activations for the WMMA GEMMs
// ---------------------------------------------------------------------------
__global__ void ln_kernel(const float* __restrict__ x,
                          const float* __restrict__ g,
                          const float* __restrict__ b,
                          _Float16* __restrict__ o)
{
    const int row = blockIdx.x;
    const float* xr = x + (size_t)row * Dm;
    __shared__ float sm[256];
    float a = 0.f;
    for (int i = threadIdx.x; i < Dm; i += 256) a += xr[i];
    sm[threadIdx.x] = a; __syncthreads();
    for (int st = 128; st; st >>= 1) {
        if ((int)threadIdx.x < st) sm[threadIdx.x] += sm[threadIdx.x + st];
        __syncthreads();
    }
    const float mean = sm[0] * (1.f / Dm);
    __syncthreads();
    a = 0.f;
    for (int i = threadIdx.x; i < Dm; i += 256) { float d = xr[i] - mean; a += d * d; }
    sm[threadIdx.x] = a; __syncthreads();
    for (int st = 128; st; st >>= 1) {
        if ((int)threadIdx.x < st) sm[threadIdx.x] += sm[threadIdx.x + st];
        __syncthreads();
    }
    const float rstd = rsqrtf(sm[0] * (1.f / Dm) + 1e-5f);
    _Float16* orow = o + (size_t)row * Dm;
    for (int i = threadIdx.x; i < Dm; i += 256)
        orow[i] = (_Float16)((xr[i] - mean) * rstd * g[i] + b[i]);
}

// ---------------------------------------------------------------------------
// Residual add: a += b
// ---------------------------------------------------------------------------
__global__ void add_kernel(float* __restrict__ a, const float* __restrict__ b, int n)
{
    int i = blockIdx.x * blockDim.x + threadIdx.x;
    if (i < n) a[i] += b[i];
}

// ---------------------------------------------------------------------------
// WMMA GEMM:  C[M,N] = A[M,K](f16) * B[K,N](f32 weights, cvt on the fly) + bias
// Each wave owns a 128x16 strip (8 M-tiles of 16x16) so every B fragment is
// reused by 8 WMMAs. 4 waves per block along N.
// A-frag layout : lane m = l&15;  e<8 -> K = 8*half+e, e>=8 -> K = 16+8*half+(e-8)
// B-frag layout : lane n = l&15;  e<8 -> K = 8*half+e, e>=8 -> K = 16+8*half+(e-8)
// C/D layout    : VGPR r -> row tile + r + 8*half, col n0 + (l&15)
// ---------------------------------------------------------------------------
template<bool RELU, typename OutT>
__global__ __launch_bounds__(128, 1)
void gemm_wmma(const _Float16* __restrict__ A,
               const float* __restrict__ B,
               const float* __restrict__ bias,
               OutT* __restrict__ C,
               int M, int N, int K)
{
    const int lane = threadIdx.x & 31;
    const int wave = threadIdx.x >> 5;
    const int n0 = (blockIdx.x * 4 + wave) * 16;
    const int m0 = blockIdx.y * 128;
    if (n0 >= N) return;                       // wave-uniform exit (EXEC stays full)
    const int half = lane >> 4;
    const int l15  = lane & 15;
    const int kbA  = half * 8;
    const int kbB  = half * 8;
    const int n    = n0 + l15;
    const bool nv  = (n < N);                  // N-tail guard (head GEMM: N=50257)
    const _Float16* Arow = A + (size_t)(m0 + l15) * K;

    v8f acc[8];
    v8f zero = {};
    #pragma unroll
    for (int mt = 0; mt < 8; ++mt) acc[mt] = zero;

    for (int k0 = 0; k0 < K; k0 += 32) {
        v16h b;
        const float* Bp = B + (size_t)(k0 + kbB) * N + n;
        #pragma unroll
        for (int e = 0; e < 16; ++e)
            b[e] = nv ? (_Float16)Bp[(size_t)(e < 8 ? e : e + 8) * N] : (_Float16)0.f;
        if (k0 + 32 < K) __builtin_prefetch(Bp + (size_t)32 * N, 0, 3);
        #pragma unroll
        for (int mt = 0; mt < 8; ++mt) {
            const _Float16* Ap = Arow + (size_t)mt * 16 * K + k0;
            v16h a;
            #pragma unroll
            for (int e = 0; e < 8; ++e) a[e]     = Ap[kbA + e];
            #pragma unroll
            for (int e = 0; e < 8; ++e) a[8 + e] = Ap[16 + kbA + e];
            acc[mt] = __builtin_amdgcn_wmma_f32_16x16x32_f16(
                false, a, false, b, (short)0, acc[mt], false, false);
        }
    }
    #pragma unroll
    for (int mt = 0; mt < 8; ++mt) {
        #pragma unroll
        for (int r = 0; r < 8; ++r) {
            if (nv) {
                float v = acc[mt][r] + bias[n];
                if (RELU) v = fmaxf(v, 0.f);
                C[(size_t)(m0 + mt * 16 + 8 * half + r) * N + n] = (OutT)v;
            }
        }
    }
}

// ---------------------------------------------------------------------------
// Flash attention (causal). Block = 2 waves; each wave owns one 16-row query
// tile; both waves share double-buffered 32x192 f16 K/V LDS tiles.
// qkv16 layout: [B, T, 3*D] f16; q at h*192, k at 768+h*192, v at 1536+h*192.
// K/V tiles are moved with GLOBAL_LOAD_ASYNC_TO_LDS_B128 (ASYNCcnt pipeline):
//   stage(i+1) -> s_wait_asynccnt<=24 (tile i done, i+1 in flight) -> barrier
//   -> compute tile i -> barrier (buffer reuse safe).
// Online softmax in f32 registers; P converted C-layout -> A-layout via LDS.
// Loop bound is block-uniform; wave 0's trailing fully-masked tile is a no-op
// under the online-softmax math (p -> 0, corr = 1).
// ---------------------------------------------------------------------------
__device__ __forceinline__ void stage_kv(const _Float16* __restrict__ kbase,
                                         const _Float16* __restrict__ vbase,
                                         int s0, int tid,
                                         _Float16 (*Kt)[DHm + 8],
                                         _Float16 (*Vt)[DHm + 8])
{
    const int srow = tid >> 1;                 // 0..31
    const int dh0  = (tid & 1) * 96;           // 0 or 96
    const size_t rs = 3 * Dm;
    const _Float16* kr = kbase + (size_t)(s0 + srow) * rs + dh0;
    const _Float16* vr = vbase + (size_t)(s0 + srow) * rs + dh0;
#if HAVE_ASYNC
    #pragma unroll
    for (int c = 0; c < 12; ++c) {             // 12 x 16B per row-half
        async_copy16B(kr + c * 8, &Kt[srow][dh0 + c * 8]);
        async_copy16B(vr + c * 8, &Vt[srow][dh0 + c * 8]);
    }
#else
    #pragma unroll 8
    for (int d = 0; d < 96; ++d) {
        Kt[srow][dh0 + d] = kr[d];
        Vt[srow][dh0 + d] = vr[d];
    }
#endif
}

__global__ __launch_bounds__(64, 1)
void attn_kernel(const _Float16* __restrict__ qkv, float* __restrict__ y)
{
    const int q0blk = blockIdx.x * 32;          // 32 query rows per block
    const int hIdx  = blockIdx.y;
    const int bIdx  = blockIdx.z;
    const int tid   = threadIdx.x;
    const int wave  = tid >> 5;
    const int lane  = tid & 31;
    const int q0    = q0blk + wave * 16;        // this wave's query tile
    const int half  = lane >> 4;
    const int l15   = lane & 15;
    const int kbA   = half * 8;
    const int kbB   = half * 8;
    const float scale = 0.07216878364870323f;   // 1/sqrt(192)
    const size_t rs = 3 * Dm;                   // qkv row stride

    const _Float16* qbase = qkv + (size_t)bIdx * Tm * rs + hIdx * DHm;
    const _Float16* kbase = qbase + Dm;
    const _Float16* vbase = qbase + 2 * Dm;

    // ---- Q fragments (6 steps of K=32 over DH=192), straight f16 loads
    v16h Qf[6];
    {
        const _Float16* qrow = qbase + (size_t)(q0 + l15) * rs;
        #pragma unroll
        for (int s = 0; s < 6; ++s) {
            #pragma unroll
            for (int e = 0; e < 16; ++e) {
                const int dh = s * 32 + ((e < 8) ? (kbA + e) : (16 + kbA + (e - 8)));
                Qf[s][e] = qrow[dh];
            }
        }
    }

    v8f zero = {};
    v8f O[12];
    #pragma unroll
    for (int t = 0; t < 12; ++t) O[t] = zero;
    float rm[8], rl[8];
    #pragma unroll
    for (int r = 0; r < 8; ++r) { rm[r] = -1e30f; rl[r] = 0.f; }

    __shared__ alignas(16) _Float16 Kt[2][32][DHm + 8];
    __shared__ alignas(16) _Float16 Vt[2][32][DHm + 8];
    __shared__ alignas(16) _Float16 Pt[2][16][40];

    const int sMax = q0blk + 31;
    stage_kv(kbase, vbase, 0, tid, Kt[0], Vt[0]);      // prime the pipeline

    int buf = 0;
    for (int s0 = 0; s0 <= sMax; s0 += 32, buf ^= 1) {
        const bool more = (s0 + 32) <= sMax;            // block-uniform
        if (more) stage_kv(kbase, vbase, s0 + 32, tid, Kt[buf ^ 1], Vt[buf ^ 1]);
#if HAVE_ASYNC
        if (more) wait_async<24>(); else wait_async<0>();
#endif
        __syncthreads();                                // tile s0 visible to both waves

        // ---- S = Q K^T, two 16x16 subtiles (cols s0..s0+15, s0+16..s0+31)
        v8f S0 = zero, S1 = zero;
        #pragma unroll
        for (int s = 0; s < 6; ++s) {
            v16h b0, b1;
            #pragma unroll
            for (int e = 0; e < 16; ++e) {
                const int dh = s * 32 + kbB + (e < 8 ? e : e + 8);
                b0[e] = Kt[buf][l15][dh];
                b1[e] = Kt[buf][l15 + 16][dh];
            }
            S0 = __builtin_amdgcn_wmma_f32_16x16x32_f16(false, Qf[s], false, b0,
                                                        (short)0, S0, false, false);
            S1 = __builtin_amdgcn_wmma_f32_16x16x32_f16(false, Qf[s], false, b1,
                                                        (short)0, S1, false, false);
        }

        // ---- scale + causal mask + online softmax (rows live per half-wave)
        #pragma unroll
        for (int r = 0; r < 8; ++r) {
            const int qrow = q0 + 8 * half + r;
            float a0 = S0[r] * scale;
            float a1 = S1[r] * scale;
            if (s0 + l15      > qrow) a0 = -1e30f;
            if (s0 + 16 + l15 > qrow) a1 = -1e30f;
            float mx = fmaxf(a0, a1);
            #pragma unroll
            for (int o = 1; o < 16; o <<= 1) mx = fmaxf(mx, __shfl_xor(mx, o, 32));
            const float nm   = fmaxf(rm[r], mx);
            const float corr = __expf(rm[r] - nm);
            const float p0   = __expf(a0 - nm);
            const float p1   = __expf(a1 - nm);
            float ps = p0 + p1;
            #pragma unroll
            for (int o = 1; o < 16; o <<= 1) ps += __shfl_xor(ps, o, 32);
            rl[r] = rl[r] * corr + ps;
            rm[r] = nm;
            #pragma unroll
            for (int t = 0; t < 12; ++t) O[t][r] *= corr;
            Pt[wave][8 * half + r][l15]      = (_Float16)p0;
            Pt[wave][8 * half + r][16 + l15] = (_Float16)p1;
        }
        __syncthreads();   // also orders Pt store->load within each wave

        // ---- P fragment in A layout (16x32)
        v16h Pf;
        #pragma unroll
        for (int e = 0; e < 16; ++e) {
            const int kk = (e < 8) ? (kbA + e) : (16 + kbA + (e - 8));
            Pf[e] = Pt[wave][l15][kk];
        }
        // ---- O += P @ V  (12 output n-tiles over DH=192)
        #pragma unroll
        for (int t = 0; t < 12; ++t) {
            v16h bv;
            #pragma unroll
            for (int e = 0; e < 16; ++e) bv[e] = Vt[buf][kbB + (e < 8 ? e : e + 8)][t * 16 + l15];
            O[t] = __builtin_amdgcn_wmma_f32_16x16x32_f16(false, Pf, false, bv,
                                                          (short)0, O[t], false, false);
        }
        __syncthreads();   // all waves done reading tile before buffer reuse
    }

    // ---- write y[b, q0+m, h*192 + dh] = O / rowsum
    float* yb = y + (size_t)bIdx * Tm * Dm + hIdx * DHm;
    #pragma unroll
    for (int r = 0; r < 8; ++r) {
        const float inv = 1.f / rl[r];
        float* yr = yb + (size_t)(q0 + 8 * half + r) * Dm;
        #pragma unroll
        for (int t = 0; t < 12; ++t) yr[t * 16 + l15] = O[t][r] * inv;
    }
}

// ---------------------------------------------------------------------------
// Host-side orchestration
// ---------------------------------------------------------------------------
extern "C" void kernel_launch(void* const* d_in, const int* in_sizes, int n_in,
                              void* d_out, int out_size, void* d_ws, size_t ws_size,
                              hipStream_t stream)
{
    (void)in_sizes; (void)n_in; (void)out_size; (void)ws_size;
    const int*   x       = (const int*)  d_in[0];
    const float* tok_emb = (const float*)d_in[1];
    const float* pos_emb = (const float*)d_in[2];
    const float* ln1_s   = (const float*)d_in[3];
    const float* ln1_b   = (const float*)d_in[4];
    const float* qkv_w   = (const float*)d_in[5];
    const float* qkv_b   = (const float*)d_in[6];
    const float* ln2_s   = (const float*)d_in[7];
    const float* ln2_b   = (const float*)d_in[8];
    const float* fc_w    = (const float*)d_in[9];
    const float* fc_b    = (const float*)d_in[10];
    const float* proj_w  = (const float*)d_in[11];
    const float* proj_b  = (const float*)d_in[12];
    const float* lnf_s   = (const float*)d_in[13];
    const float* lnf_b   = (const float*)d_in[14];
    const float* head_w  = (const float*)d_in[15];
    const float* head_b  = (const float*)d_in[16];
    float* out = (float*)d_out;

    char* ws = (char*)d_ws;
    size_t off = 0;
    float*     h      = (float*)(ws + off);     off += (size_t)ROWS * Dm * 4;       // 12.6 MB
    float*     yb     = (float*)(ws + off);     off += (size_t)ROWS * Dm * 4;       // 12.6 MB
    _Float16*  qkv16  = (_Float16*)(ws + off);  off += (size_t)ROWS * 3 * Dm * 2;   // 18.9 MB
    _Float16*  ln16   = (_Float16*)(ws + off);  off += (size_t)ROWS * Dm * 2;       //  6.3 MB
    _Float16*  mlp16  = (_Float16*)(ws + off);  off += (size_t)ROWS * 4 * Dm * 2;   // 25.2 MB

    const dim3 gemmBlk(128);                              // 4 waves
    const dim3 gQKV((3 * Dm + 63) / 64, ROWS / 128);      // N=2304
    const dim3 gFC((4 * Dm + 63) / 64, ROWS / 128);       // N=3072
    const dim3 gPRJ((Dm + 63) / 64, ROWS / 128);          // N=768
    const dim3 gHEAD((Vm + 63) / 64, ROWS / 128);         // N=50257
    const int  nElem = ROWS * Dm;

    embed_kernel<<<ROWS, 256, 0, stream>>>(x, tok_emb, pos_emb, h);

    for (int l = 0; l < Lm; ++l) {
        ln_kernel<<<ROWS, 256, 0, stream>>>(h, ln1_s + (size_t)l * Dm,
                                            ln1_b + (size_t)l * Dm, ln16);
        gemm_wmma<false, _Float16><<<gQKV, gemmBlk, 0, stream>>>(
            ln16, qkv_w + (size_t)l * Dm * 3 * Dm, qkv_b + (size_t)l * 3 * Dm,
            qkv16, ROWS, 3 * Dm, Dm);
        attn_kernel<<<dim3(Tm / 32, Hm, Bm), 64, 0, stream>>>(qkv16, yb);
        add_kernel<<<(nElem + 255) / 256, 256, 0, stream>>>(h, yb, nElem);

        ln_kernel<<<ROWS, 256, 0, stream>>>(h, ln2_s + (size_t)l * Dm,
                                            ln2_b + (size_t)l * Dm, ln16);
        gemm_wmma<true, _Float16><<<gFC, gemmBlk, 0, stream>>>(
            ln16, fc_w + (size_t)l * Dm * 4 * Dm, fc_b + (size_t)l * 4 * Dm,
            mlp16, ROWS, 4 * Dm, Dm);
        gemm_wmma<false, float><<<gPRJ, gemmBlk, 0, stream>>>(
            mlp16, proj_w + (size_t)l * 4 * Dm * Dm, proj_b + (size_t)l * Dm,
            yb, ROWS, Dm, 4 * Dm);
        add_kernel<<<(nElem + 255) / 256, 256, 0, stream>>>(h, yb, nElem);
    }

    ln_kernel<<<ROWS, 256, 0, stream>>>(h, lnf_s, lnf_b, ln16);
    gemm_wmma<false, float><<<gHEAD, gemmBlk, 0, stream>>>(
        ln16, head_w, head_b, out, ROWS, Vm, Dm);
}
